// binary_conv_67516885893425
// MI455X (gfx1250) — matrix operand order repaired and checked by reading, not compiled, not run on hardware
//
#include <hip/hip_runtime.h>

typedef __attribute__((ext_vector_type(16))) _Float16 v16h;
typedef __attribute__((ext_vector_type(8)))  _Float16 v8h;
typedef __attribute__((ext_vector_type(8)))  float    v8f;

// Problem geometry (from reference)
constexpr int LEVELS = 2;
constexpr int Bn = 16, H = 32, Wd = 32, Cc = 128, Fn = 128, Kk = 3;
constexpr int HO = H - Kk + 1, WO = Wd - Kk + 1;       // 30, 30
constexpr int WIN  = Kk * Kk * Cc;                      // 1152
constexpr int Mtot = Bn * HO * WO;                      // 14400 (divisible by 16)
constexpr int NKC  = WIN / 32;                          // 36 K-chunks of 32
constexpr int NTERMS = 16;                              // LUT terms per level

// ---------------------------------------------------------------------------
// Kernel 1: binarize weights, scale by |gamma|, store f16 transposed:
//   wbinT[t][f][n] = |gamma| * sign(w[t][n][f])   (contiguous K runs per column)
// ---------------------------------------------------------------------------
__global__ void prep_w(const float* __restrict__ w, const float* __restrict__ gamma,
                       _Float16* __restrict__ wbinT) {
    int idx = blockIdx.x * 256 + threadIdx.x;
    if (idx >= 32 * WIN * Fn) return;
    int f = idx % Fn;
    int n = (idx / Fn) % WIN;
    int t = idx / (Fn * WIN);
    float ga = fabsf(gamma[0]);
    float wv = w[idx];                       // flat [32][1152][128] matches reference reshape
    float s  = (wv > 0.f) ? ga : ((wv < 0.f) ? -ga : 0.f);
    wbinT[((size_t)(t * Fn + f)) * WIN + n] = (_Float16)s;
}

// ---------------------------------------------------------------------------
// Kernel 2: per (level, pixel, n): magnitude |p| (f16) and 4-bit term code (u8).
// code = 8*[p<0] | 4*[s0<0] | 2*[s1<0] | [s2<0]   (term index within level)
// ---------------------------------------------------------------------------
__global__ void prep_act(const float* __restrict__ x,
                         const int* __restrict__ rm0, const int* __restrict__ rm1,
                         const int* __restrict__ rm2,
                         _Float16* __restrict__ vals, unsigned char* __restrict__ codes) {
    long long idx = (long long)blockIdx.x * 256 + threadIdx.x;
    const long long total = (long long)LEVELS * Mtot * WIN;
    if (idx >= total) return;
    int n = (int)(idx % WIN);
    int m = (int)((idx / WIN) % Mtot);
    int l = (int)(idx / ((long long)WIN * Mtot));
    int wo = m % WO;
    int ho = (m / WO) % HO;
    int b  = m / (WO * HO);

    auto fetch = [&](int nn) -> float {
        int ch  = nn % Cc;
        int blk = nn / Cc;          // (kr*3 + kq), channel fastest (TF patch order)
        int kq  = blk % Kk;
        int kr  = blk / Kk;
        return x[(((((long long)l * Bn + b) * H + (ho + kr)) * Wd + (wo + kq)) * Cc) + ch];
    };

    float p  = fetch(n);
    float s0 = fetch(rm0[n]);
    float s1 = fetch(rm1[n]);
    float s2 = fetch(rm2[n]);
    unsigned code = ((p  < 0.f) ? 8u : 0u) | ((s0 < 0.f) ? 4u : 0u) |
                    ((s1 < 0.f) ? 2u : 0u) | ((s2 < 0.f) ? 1u : 0u);
    vals[idx]  = (_Float16)fabsf(p);
    codes[idx] = (unsigned char)code;
}

// ---------------------------------------------------------------------------
// TDM: DMA one 2D slab (rows x rowbytes, stride stridebytes) into LDS.
// Descriptor layout per CDNA5 ISA (groups 0/1; 2D tensor,
// groups 2/3 zero). data_size = 4 bytes, dims in dword units.
// ---------------------------------------------------------------------------
__device__ inline void tdm_load_2d(const void* src, unsigned lds_off,
                                   unsigned rows, unsigned rowbytes,
                                   unsigned stridebytes) {
    typedef __attribute__((ext_vector_type(4))) unsigned int u32x4;
    typedef __attribute__((ext_vector_type(4))) int          i32x4;
    typedef __attribute__((ext_vector_type(8))) int          i32x8;

    unsigned long long ga = (unsigned long long)src;
    u32x4 g0;
    g0[0] = 1u;                                            // count=1, user descriptor
    g0[1] = lds_off;                                       // LDS byte address
    g0[2] = (unsigned)ga;                                  // global_addr[31:0]
    g0[3] = (unsigned)((ga >> 32) & 0x01FFFFFFu)           // global_addr[56:32]
          | (2u << 30);                                    // type = 2 ("image")

    const unsigned dim0  = stridebytes >> 2;               // tensor_dim0 (dwords)
    const unsigned dim1  = rows;                           // tensor_dim1
    const unsigned tile0 = rowbytes >> 2;                  // tile_dim0 (dwords)
    const unsigned tile1 = rows;                           // tile_dim1
    const unsigned long long str0 = (unsigned long long)(stridebytes >> 2);

    i32x8 g1;
    g1[0] = (int)(2u << 16);                               // data_size=2 -> 4B; mask=0
    g1[1] = (int)((dim0 & 0xFFFFu) << 16);                 // tensor_dim0[15:0] @ [31:16]
    g1[2] = (int)(((dim0 >> 16) & 0xFFFFu) |               // tensor_dim0[31:16]
                  ((dim1 & 0xFFFFu) << 16));               // tensor_dim1[15:0]
    g1[3] = (int)(((dim1 >> 16) & 0xFFFFu) |               // tensor_dim1[31:16]
                  ((tile0 & 0xFFFFu) << 16));              // tile_dim0
    g1[4] = (int)(tile1 & 0xFFFFu);                        // tile_dim1; tile_dim2=0
    g1[5] = (int)(str0 & 0xFFFFFFFFull);                   // tensor_dim0_stride[31:0]
    g1[6] = (int)((str0 >> 32) & 0xFFFFull);               // stride[47:32]; dim1_stride=0
    g1[7] = 0;

    i32x4 z4 = {0, 0, 0, 0};
#if __clang_major__ >= 23
    i32x8 z8 = {0, 0, 0, 0, 0, 0, 0, 0};
    __builtin_amdgcn_tensor_load_to_lds(g0, g1, z4, z4, z8, 0);
#else
    __builtin_amdgcn_tensor_load_to_lds(g0, g1, z4, z4, 0);
#endif
}

// ---------------------------------------------------------------------------
// Kernel 3: WMMA GEMM. Wave computes a 16(M)x64(N) output tile.
// 288 flattened stages (level x kchunk x 4-term group); TDM double-buffers the
// [4 terms][128 F][32 K] f16 weight slab in LDS while WMMAs consume the other.
// ---------------------------------------------------------------------------
__global__ __launch_bounds__(256) void conv_wmma(
    const _Float16* __restrict__ wbinT, const _Float16* __restrict__ vals,
    const unsigned char* __restrict__ codes, float* __restrict__ out) {

    constexpr int TS    = 4;                      // terms per stage
    constexpr int SLABH = TS * Fn;                // 512 rows of 64B
    constexpr int SLABE = SLABH * 32;             // 16384 f16 per slab (32 KB)
    constexpr int NST   = LEVELS * NKC * (NTERMS / TS);   // 288 stages
    __shared__ _Float16 ldsW[2 * SLABE];          // 64 KB, double buffered

    const int tid   = threadIdx.x;
    const int lane  = tid & 31;
    const int wave  = tid >> 5;
    const int wid   = blockIdx.x * 8 + wave;      // 0..1799
    const int mtile = wid >> 1;                   // 0..899
    const int nhalf = wid & 1;                    // which 64-wide N half
    const int mrow  = lane & 15;                  // A row / B,C column
    const int kA    = (lane < 16) ? 0 : 8;        // A-frag K base (ISA 16-bit A 16x32)
    const int kB    = (lane < 16) ? 0 : 8;        // B-frag K base (ISA 16-bit B 32x16)

    // LDS byte offset of the slab (generic->LDS mapping uses addr[31:0])
    const unsigned ldsbase = (unsigned)(unsigned long long)(uintptr_t)&ldsW[0];

    auto issue = [&](int s) {
        const int l  = s / (NKC * (NTERMS / TS));
        const int kc = (s / (NTERMS / TS)) % NKC;
        const int tt = (s % (NTERMS / TS)) * TS;
        const _Float16* src =
            wbinT + ((long long)((l * NTERMS + tt) * Fn)) * WIN + kc * 32;
        tdm_load_2d(src, ldsbase + (unsigned)((s & 1) * SLABE * 2),
                    SLABH, 64u, (unsigned)(WIN * 2));
    };

    v8f acc[4] = {};

    if (wave == 0) issue(0);                      // prime the pipeline

    for (int s = 0; s < NST; ++s) {
        const int l  = s / (NKC * (NTERMS / TS));
        const int kc = (s / (NTERMS / TS)) % NKC;
        const int tt = (s % (NTERMS / TS)) * TS;

        // ---- per-lane A data: |p| values + term codes (two 16B / two 8B loads)
        const long long rowoff =
            ((long long)(l * Mtot + mtile * 16 + mrow)) * WIN + kc * 32;
        const _Float16*      vp = vals  + rowoff;
        const unsigned char* cp = codes + rowoff;
        union { v16h v; v8h h2[2]; _Float16 h[16]; } va;
        va.h2[0] = *(const v8h*)(vp + kA);
        va.h2[1] = *(const v8h*)(vp + kA + 16);
        unsigned long long cc0 = *(const unsigned long long*)(cp + kA);
        unsigned long long cc1 = *(const unsigned long long*)(cp + kA + 16);
        if ((s & 3) == 3) __builtin_prefetch(vp + 32, 0, 1);   // next K-chunk row

        // ---- double-buffer control: issue next slab, wait for current
        if (wave == 0) {
            if (s + 1 < NST) {
                issue(s + 1);
                __builtin_amdgcn_s_wait_tensorcnt(1);   // stage s slab complete
            } else {
                __builtin_amdgcn_s_wait_tensorcnt(0);
            }
        }
        __syncthreads();                                 // publish slab s

        const _Float16* lw = &ldsW[(s & 1) * SLABE];
        #pragma unroll
        for (int t4 = 0; t4 < TS; ++t4) {
            const unsigned long long code_t = (unsigned long long)(tt + t4);
            // ---- mask A fragment: keep |p| only where code == term
            union { v16h v; _Float16 h[16]; } at;
            #pragma unroll
            for (int e = 0; e < 8; ++e) {
                at.h[e]     = (((cc0 >> (8 * e)) & 0xFFull) == code_t)
                              ? va.h[e]     : (_Float16)0.f;
                at.h[e + 8] = (((cc1 >> (8 * e)) & 0xFFull) == code_t)
                              ? va.h[e + 8] : (_Float16)0.f;
            }
            #pragma unroll
            for (int nt = 0; nt < 4; ++nt) {
                const int fcol = nhalf * 64 + nt * 16 + mrow;
                union { v16h v; v8h h2[2]; } bf;
                bf.h2[0] = *(const v8h*)&lw[(t4 * Fn + fcol) * 32 + kB];
                bf.h2[1] = *(const v8h*)&lw[(t4 * Fn + fcol) * 32 + kB + 16];
                acc[nt] = __builtin_amdgcn_wmma_f32_16x16x32_f16(
                    false, at.v, false, bf.v, (short)0, acc[nt], false, false);
            }
        }
        __syncthreads();     // slab (s&1) free for reuse at stage s+2
    }

    // ---- store C (ISA f32 C/D layout: vgpr r -> M = r (+8 for upper half-wave))
    #pragma unroll
    for (int nt = 0; nt < 4; ++nt) {
        const int col = nhalf * 64 + nt * 16 + mrow;
        #pragma unroll
        for (int r = 0; r < 8; ++r) {
            int mm = mtile * 16 + r + ((lane < 16) ? 0 : 8);
            out[(long long)mm * Fn + col] = acc[nt][r];
        }
    }
}

// ---------------------------------------------------------------------------
extern "C" void kernel_launch(void* const* d_in, const int* in_sizes, int n_in,
                              void* d_out, int out_size, void* d_ws, size_t ws_size,
                              hipStream_t stream) {
    const float* x     = (const float*)d_in[0];   // [2,16,32,32,128] f32
    const float* w     = (const float*)d_in[1];   // [32,3,3,128,128] f32
    const float* gamma = (const float*)d_in[2];   // [1] f32
    const int*   rm0   = (const int*)d_in[3];     // [1152] i32
    const int*   rm1   = (const int*)d_in[4];
    const int*   rm2   = (const int*)d_in[5];
    float* out = (float*)d_out;                   // [16,30,30,128] f32

    char* ws = (char*)d_ws;
    _Float16* wbinT = (_Float16*)ws;                                  // 9,437,184 B
    size_t offV = (size_t)32 * WIN * Fn * sizeof(_Float16);
    _Float16* vals = (_Float16*)(ws + offV);                          // 66,355,200 B
    size_t offC = offV + (size_t)LEVELS * Mtot * WIN * sizeof(_Float16);
    unsigned char* codes = (unsigned char*)(ws + offC);               // 33,177,600 B

    // 1) binarize + transpose weights
    int n1 = 32 * WIN * Fn;
    prep_w<<<(n1 + 255) / 256, 256, 0, stream>>>(w, gamma, wbinT);

    // 2) activation magnitudes + LUT codes
    long long n2 = (long long)LEVELS * Mtot * WIN;
    prep_act<<<(unsigned)((n2 + 255) / 256), 256, 0, stream>>>(x, rm0, rm1, rm2,
                                                               vals, codes);

    // 3) masked-LUT WMMA GEMM with TDM double-buffered weights
    conv_wmma<<<(Mtot / 16) * 2 / 8, 256, 0, stream>>>(wbinT, vals, codes, out);
}
